// RNNSearch_24206435680497
// MI455X (gfx1250) — matrix operand order repaired and checked by reading, not compiled, not run on hardware
//
#include <hip/hip_runtime.h>
#include <math.h>

// ---------------- types ----------------
typedef __attribute__((ext_vector_type(16))) __bf16 bf16x16;
typedef __attribute__((ext_vector_type(8)))  __bf16 bf16x8;
typedef __attribute__((ext_vector_type(8)))  float  f32x8;

// ---------------- problem constants ----------------
constexpr int kB    = 64;     // batch
constexpr int kS    = 64;     // src length
constexpr int kT    = 64;     // trg length
constexpr int kV    = 32000;  // vocab
constexpr int kEIN  = 512;    // encoder input dim
constexpr int kEH   = 512;    // encoder hidden
constexpr int kDH   = 1024;   // decoder hidden
constexpr int kCTX  = 1024;   // 2*kEH
constexpr int kNATT = 512;
constexpr int kNRD  = 512;

constexpr int GF_BIAS = 1;
constexpr int GF_ACC  = 2;

// GEMM register blocking: each wave computes MT x NT 16x16 tiles
constexpr int GMT = 4;   // m-tiles per wave  -> 64 rows per block row
constexpr int GNT = 2;   // n-tiles per wave  -> 4 waves * 32 = 128 cols per block

// ---------------- helpers ----------------
__device__ __forceinline__ __bf16 f2bf(float f) {
  unsigned u = __float_as_uint(f);
  u += 0x7FFFu + ((u >> 16) & 1u);               // round-to-nearest-even
  unsigned short s = (unsigned short)(u >> 16);
  __bf16 r; __builtin_memcpy(&r, &s, sizeof(r));
  return r;
}
__device__ __forceinline__ float sigm(float x) { return 1.0f / (1.0f + expf(-x)); }

// ---------------- WMMA GEMM: C[M,N] = A[M,K] * W[N,K]^T (+bias)(+acc) ----------------
// A, W bf16 row-major with leading dims lda/ldw; C fp32 row-major ldc.
// Wave computes a 64x32 output patch (4 m-tiles x 2 n-tiles) so each A fragment
// feeds 2 WMMAs and each B fragment feeds 4 WMMAs: 8 WMMAs per 12 b128 loads.
// Requirements (all call sites satisfy): K % 32 == 0, M % 64 == 0, N % 128 == 0.
__global__ __launch_bounds__(128)
void rnnsearch_gemm_wmma(const __bf16* __restrict__ A, int lda,
                         const __bf16* __restrict__ W, int ldw,
                         const float* __restrict__ bias,
                         float* __restrict__ C, int ldc,
                         int M, int N, int K, int flags)
{
  const int lane  = threadIdx.x & 31;
  const int wave  = threadIdx.x >> 5;
  const int n0    = (blockIdx.x * 4 + wave) * (GNT * 16);
  const int m0    = blockIdx.y * (GMT * 16);
  if (n0 + GNT * 16 > N || m0 + GMT * 16 > M) return;

  const int l15   = lane & 15;
  const int khalf = lane >> 4;       // lanes 16-31 hold upper K-halves

  // A fragment rows: lane l15 = M-row; element j: k = j + 16*khalf
  const __bf16* pa[GMT];
#pragma unroll
  for (int i = 0; i < GMT; ++i)
    pa[i] = A + (size_t)(m0 + i * 16 + l15) * lda + khalf * 16;
  // B fragment: lane l15 = N-col (W row); element j: k = j + 16*khalf
  const __bf16* pw[GNT];
#pragma unroll
  for (int i = 0; i < GNT; ++i)
    pw[i] = W + (size_t)(n0 + i * 16 + l15) * ldw + khalf * 16;

  f32x8 acc[GMT][GNT];
#pragma unroll
  for (int i = 0; i < GMT; ++i)
#pragma unroll
    for (int j = 0; j < GNT; ++j)
      acc[i][j] = (f32x8)(0.0f);

  for (int k0 = 0; k0 < K; k0 += 32) {
    bf16x16 afr[GMT];
#pragma unroll
    for (int i = 0; i < GMT; ++i) {
      bf16x8 lo = *(const bf16x8*)(pa[i] + k0);        // k = k0 + 16*khalf + 0..7
      bf16x8 hi = *(const bf16x8*)(pa[i] + k0 + 8);    // k = k0 + 16*khalf + 8..15
#pragma unroll
      for (int j = 0; j < 8; ++j) { afr[i][j] = lo[j]; afr[i][j + 8] = hi[j]; }
    }
    bf16x16 bfr[GNT];
#pragma unroll
    for (int i = 0; i < GNT; ++i)
      bfr[i] = *(const bf16x16*)(pw[i] + k0);          // k = k0 + 16*khalf + 0..15

    // prefetch next K-chunk of the (large) weight panel: global_prefetch_b8
    if (k0 + 32 < K) {
#pragma unroll
      for (int i = 0; i < GNT; ++i) __builtin_prefetch(pw[i] + k0 + 32, 0, 3);
    }

#pragma unroll
    for (int i = 0; i < GMT; ++i)
#pragma unroll
      for (int j = 0; j < GNT; ++j)
        acc[i][j] = __builtin_amdgcn_wmma_f32_16x16x32_bf16(
            /*neg_a=*/false, afr[i], /*neg_b=*/false, bfr[j],
            /*c_mod=*/(short)0, acc[i][j], /*reuse_a=*/false, /*reuse_b=*/false);
  }

#pragma unroll
  for (int j = 0; j < GNT; ++j) {
    const int ocol = n0 + j * 16 + l15;
    float bv = (flags & GF_BIAS) ? bias[ocol] : 0.0f;
#pragma unroll
    for (int i = 0; i < GMT; ++i) {
      const int orow0 = m0 + i * 16 + khalf * 8;       // VGPR r -> M = r + 8*khalf
#pragma unroll
      for (int r = 0; r < 8; ++r) {
        size_t idx = (size_t)(orow0 + r) * ldc + ocol;
        float v = acc[i][j][r] + bv;
        if (flags & GF_ACC) v += C[idx];
        C[idx] = v;
      }
    }
  }
}

// ---------------- elementwise / small kernels ----------------
__global__ __launch_bounds__(256)
void rnnsearch_f2bf(const float* __restrict__ in, __bf16* __restrict__ out, int n)
{
  int i = blockIdx.x * blockDim.x + threadIdx.x;
  if (i < n) out[i] = f2bf(in[i]);
}

__global__ __launch_bounds__(256)
void rnnsearch_zero_state(float* __restrict__ h, __bf16* __restrict__ hb, int n)
{
  int i = blockIdx.x * blockDim.x + threadIdx.x;
  if (i < n) { h[i] = 0.0f; hb[i] = f2bf(0.0f); }
}

// xs[s][b][e] = src_emb[src[b][s]][e]   (bf16 only; used solely as GEMM input)
__global__ __launch_bounds__(256)
void rnnsearch_gather_src(const int* __restrict__ src, const float* __restrict__ emb,
                          __bf16* __restrict__ xs_bf)
{
  int i = blockIdx.x * blockDim.x + threadIdx.x;
  if (i >= kS * kB * kEIN) return;
  int e = i % kEIN; int r = i / kEIN; int b = r % kB; int s = r / kB;
  xs_bf[i] = f2bf(emb[(size_t)src[b * kS + s] * kEIN + e]);
}

// temb[b][t][e] = trg_emb[f_trg[b][t]][e]
__global__ __launch_bounds__(256)
void rnnsearch_gather_trg(const int* __restrict__ ftrg, const float* __restrict__ emb,
                          __bf16* __restrict__ temb_bf)
{
  int i = blockIdx.x * blockDim.x + threadIdx.x;
  if (i >= kB * kT * 512) return;
  int e = i % 512; int r = i / 512;       // r = b*kT + t
  temb_bf[i] = f2bf(emb[(size_t)ftrg[r] * 512 + e]);
}

// encoder GRU gates + mask freeze + masked output into enc_ctx slice
__global__ __launch_bounds__(256)
void rnnsearch_gru_enc_pw(const float* __restrict__ gi, const float* __restrict__ gh,
                          const float* __restrict__ smask, int s,
                          float* __restrict__ h, __bf16* __restrict__ hbf,
                          float* __restrict__ enc_out, int out_bstride)
{
  int i = blockIdx.x * blockDim.x + threadIdx.x;
  if (i >= kB * kEH) return;
  int b = i / kEH, j = i % kEH;
  const float* gib = gi + (size_t)b * 3 * kEH;
  const float* ghb = gh + (size_t)b * 3 * kEH;
  float r = sigm(gib[j] + ghb[j]);
  float z = sigm(gib[kEH + j] + ghb[kEH + j]);
  float n = tanhf(gib[2 * kEH + j] + r * ghb[2 * kEH + j]);
  float hv = h[i];
  float hn = (1.0f - z) * n + z * hv;
  float m  = smask[b * kS + s];
  hv = m * hn + (1.0f - m) * hv;
  h[i] = hv; hbf[i] = f2bf(hv);
  enc_out[(size_t)b * out_bstride + j] = m * hv;
}

// decoder GRU gates (no mask); separate h_in/h_out
__global__ __launch_bounds__(256)
void rnnsearch_gru_dec_pw(const float* __restrict__ gi, const float* __restrict__ gh,
                          const float* __restrict__ h_in,
                          float* __restrict__ h_out, __bf16* __restrict__ h_out_bf, int H)
{
  int i = blockIdx.x * blockDim.x + threadIdx.x;
  if (i >= kB * H) return;
  int b = i / H, j = i % H;
  const float* gib = gi + (size_t)b * 3 * H;
  const float* ghb = gh + (size_t)b * 3 * H;
  float r = sigm(gib[j] + ghb[j]);
  float z = sigm(gib[H + j] + ghb[H + j]);
  float n = tanhf(gib[2 * H + j] + r * ghb[2 * H + j]);
  float hv = h_in[i];
  float o  = (1.0f - z) * n + z * hv;
  h_out[i] = o; h_out_bf[i] = f2bf(o);
}

// avg[b][d] = sum_s enc_ctx[b][s][d] / sum_s smask[b][s]   (bf16 for init GEMM)
__global__ __launch_bounds__(256)
void rnnsearch_enc_avg(const float* __restrict__ enc_ctx, const float* __restrict__ smask,
                       __bf16* __restrict__ avg_bf)
{
  int i = blockIdx.x * blockDim.x + threadIdx.x;
  if (i >= kB * kCTX) return;
  int b = i / kCTX, d = i % kCTX;
  float ms = 0.0f, sum = 0.0f;
  for (int s = 0; s < kS; ++s) {
    ms  += smask[b * kS + s];
    sum += enc_ctx[((size_t)b * kS + s) * kCTX + d];
  }
  avg_bf[i] = f2bf(sum / ms);
}

__global__ __launch_bounds__(256)
void rnnsearch_tanh_bf(float* __restrict__ x, __bf16* __restrict__ xb, int n)
{
  int i = blockIdx.x * blockDim.x + threadIdx.x;
  if (i >= n) return;
  float v = tanhf(x[i]);
  x[i] = v; xb[i] = f2bf(v);
}

// logit[b][s] = tanh(attn_ctx[b,s,:] + hs[b,:]) . a2oW + a2ob, masked to -inf
__global__ __launch_bounds__(256)
void rnnsearch_attn_logits(const float* __restrict__ attn_ctx, const float* __restrict__ hs,
                           const float* __restrict__ a2ow, const float* __restrict__ a2ob,
                           const float* __restrict__ smask, float* __restrict__ logit)
{
  int wid  = blockIdx.x * 8 + (threadIdx.x >> 5);
  int lane = threadIdx.x & 31;
  int b = wid / kS, s = wid % kS;
  const float* ac = attn_ctx + (size_t)wid * kNATT;
  const float* hb = hs + (size_t)b * kNATT;
  float acc = 0.0f;
  for (int j = lane; j < kNATT; j += 32)
    acc += tanhf(ac[j] + hb[j]) * a2ow[j];
  for (int o = 16; o; o >>= 1) acc += __shfl_xor(acc, o, 32);
  if (lane == 0) {
    float v = acc + a2ob[0];
    logit[wid] = (smask[b * kS + s] > 0.0f) ? v : -INFINITY;
  }
}

__global__ __launch_bounds__(64)
void rnnsearch_softmax(const float* __restrict__ logit, float* __restrict__ alpha)
{
  __shared__ float red[64];
  int b = blockIdx.x, s = threadIdx.x;
  float v = logit[b * kS + s];
  red[s] = v; __syncthreads();
  for (int o = 32; o; o >>= 1) { if (s < o) red[s] = fmaxf(red[s], red[s + o]); __syncthreads(); }
  float mx = red[0]; __syncthreads();
  float e = (v == -INFINITY) ? 0.0f : expf(v - mx);
  red[s] = e; __syncthreads();
  for (int o = 32; o; o >>= 1) { if (s < o) red[s] += red[s + o]; __syncthreads(); }
  alpha[b * kS + s] = e / red[0];
}

// ctx[b][d] = sum_s alpha[b][s] * enc_ctx[b][s][d]
__global__ __launch_bounds__(256)
void rnnsearch_attn_context(const float* __restrict__ alpha, const float* __restrict__ enc_ctx,
                            float* __restrict__ ctx, __bf16* __restrict__ ctx_bf)
{
  int i = blockIdx.x * blockDim.x + threadIdx.x;
  if (i >= kB * kCTX) return;
  int b = i / kCTX, d = i % kCTX;
  float acc = 0.0f;
  for (int s = 0; s < kS; ++s)
    acc += alpha[b * kS + s] * enc_ctx[((size_t)b * kS + s) * kCTX + d];
  ctx[i] = acc; ctx_bf[i] = f2bf(acc);
}

// per-row log-softmax NLL at tgt + argmax prediction
__global__ __launch_bounds__(256)
void rnnsearch_vocab_loss(const float* __restrict__ logits, const int* __restrict__ ftrg,
                          const float* __restrict__ tmask, int t,
                          float* __restrict__ loss_acc, float* __restrict__ out_pred)
{
  __shared__ float smax[256];
  __shared__ int   sidx[256];
  __shared__ float ssum[256];
  int b = blockIdx.x, tid = threadIdx.x;
  const float* row = logits + (size_t)b * kV;

  float best = -INFINITY; int bi = 0;
  for (int j = tid; j < kV; j += 256) {
    float v = row[j];
    if (v > best) { best = v; bi = j; }
  }
  smax[tid] = best; sidx[tid] = bi; __syncthreads();
  for (int o = 128; o; o >>= 1) {
    if (tid < o) {
      if (smax[tid + o] > smax[tid] ||
          (smax[tid + o] == smax[tid] && sidx[tid + o] < sidx[tid])) {
        smax[tid] = smax[tid + o]; sidx[tid] = sidx[tid + o];
      }
    }
    __syncthreads();
  }
  float mx = smax[0]; int amax = sidx[0]; __syncthreads();

  float se = 0.0f;
  for (int j = tid; j < kV; j += 256) se += expf(row[j] - mx);
  ssum[tid] = se; __syncthreads();
  for (int o = 128; o; o >>= 1) { if (tid < o) ssum[tid] += ssum[tid + o]; __syncthreads(); }

  if (tid == 0) {
    int   tgt = ftrg[b * kT + t + 1];
    float tm  = tmask[b * kT + t + 1];
    float lsm = row[tgt] - mx - logf(ssum[0]);
    loss_acc[b] += -lsm * tm;
    out_pred[b * kT + t + 1] = (float)amax;
  }
}

__global__ void rnnsearch_init_out(float* __restrict__ loss_acc, float* __restrict__ out)
{
  int tid = threadIdx.x;          // 64 threads
  loss_acc[tid] = 0.0f;
  out[2 + tid * kT] = 0.0f;       // outputs[:,0] = 0
}

__global__ __launch_bounds__(64)
void rnnsearch_finalize(const float* __restrict__ loss_acc, const float* __restrict__ tmask,
                        float* __restrict__ out)
{
  __shared__ float red[64];
  int tid = threadIdx.x;
  red[tid] = loss_acc[tid]; __syncthreads();
  for (int o = 32; o; o >>= 1) { if (tid < o) red[tid] += red[tid + o]; __syncthreads(); }
  float total = red[0]; __syncthreads();
  float ms = 0.0f;
  for (int i = tid; i < kB * kT; i += 64) { if ((i % kT) >= 1) ms += tmask[i]; }
  red[tid] = ms; __syncthreads();
  for (int o = 32; o; o >>= 1) { if (tid < o) red[tid] += red[tid + o]; __syncthreads(); }
  if (tid == 0) { out[0] = total / (float)kB; out[1] = total / red[0]; }
}

// ---------------- host orchestration ----------------
extern "C" void kernel_launch(void* const* d_in, const int* in_sizes, int n_in,
                              void* d_out, int out_size, void* d_ws, size_t ws_size,
                              hipStream_t stream)
{
  (void)in_sizes; (void)n_in; (void)out_size; (void)ws_size;

  const int*   src   = (const int*)d_in[0];
  const float* smask = (const float*)d_in[1];
  const int*   ftrg  = (const int*)d_in[2];
  const float* tmask = (const float*)d_in[3];
  const float* P[34];
  for (int i = 0; i < 34; ++i) P[i] = (const float*)d_in[4 + i];
  // param indices (setup_inputs dict order)
  // 0 src_emb 1 trg_emb 2 ef_Wih 3 ef_Whh 4 ef_bih 5 ef_bhh 6 eb_Wih 7 eb_Whh 8 eb_bih 9 eb_bhh
  // 10 g1_Wih 11 g1_Whh 12 g1_bih 13 g1_bhh 14 g2_Wih 15 g2_Whh 16 g2_bih 17 g2_bhh
  // 18 h2s_W 19 h2s_b 20 s2s_W 21 s2s_b 22 a2o_W 23 a2o_b 24 e2o_W 25 e2o_b
  // 26 h2o_W 27 h2o_b 28 c2o_W 29 c2o_b 30 aff_W 31 aff_b 32 init_W 33 init_b
  float* out = (float*)d_out;

  char* wsp = (char*)d_ws;
  auto afl = [&](size_t n) -> float*  { float*  p = (float*) wsp; wsp += ((n * 4 + 255) & ~(size_t)255); return p; };
  auto abf = [&](size_t n) -> __bf16* { __bf16* p = (__bf16*)wsp; wsp += ((n * 2 + 255) & ~(size_t)255); return p; };

  // bf16 weight copies (L2-resident working set)
  __bf16* w_ef_ih = abf((size_t)3 * kEH * kEIN);
  __bf16* w_ef_hh = abf((size_t)3 * kEH * kEH);
  __bf16* w_eb_ih = abf((size_t)3 * kEH * kEIN);
  __bf16* w_eb_hh = abf((size_t)3 * kEH * kEH);
  __bf16* w_g1_ih = abf((size_t)3 * kDH * 512);
  __bf16* w_g1_hh = abf((size_t)3 * kDH * kDH);
  __bf16* w_g2_ih = abf((size_t)3 * kDH * kCTX);
  __bf16* w_g2_hh = abf((size_t)3 * kDH * kDH);
  __bf16* w_h2s   = abf((size_t)kNATT * kDH);
  __bf16* w_s2s   = abf((size_t)kNATT * kCTX);
  __bf16* w_e2o   = abf((size_t)kNRD * 512);
  __bf16* w_h2o   = abf((size_t)kNRD * kDH);
  __bf16* w_c2o   = abf((size_t)kNRD * kCTX);
  __bf16* w_aff   = abf((size_t)kV * kNRD);
  __bf16* w_init  = abf((size_t)kDH * kCTX);

  // activations
  __bf16* xs_bf      = abf((size_t)kS * kB * kEIN);
  __bf16* temb_bf    = abf((size_t)kB * kT * 512);
  float*  gi_fw      = afl((size_t)kS * kB * 3 * kEH);
  float*  gi_bk      = afl((size_t)kS * kB * 3 * kEH);
  float*  gh_enc     = afl((size_t)kB * 3 * kEH);
  float*  h_enc      = afl((size_t)kB * kEH);
  __bf16* h_enc_bf   = abf((size_t)kB * kEH);
  float*  enc_ctx    = afl((size_t)kB * kS * kCTX);
  __bf16* enc_ctx_bf = abf((size_t)kB * kS * kCTX);
  __bf16* avg_bf     = abf((size_t)kB * kCTX);
  float*  hid        = afl((size_t)kB * kDH);
  __bf16* hid_bf     = abf((size_t)kB * kDH);
  float*  h1         = afl((size_t)kB * kDH);
  __bf16* h1_bf      = abf((size_t)kB * kDH);
  float*  attn_ctx   = afl((size_t)kB * kS * kNATT);
  float*  gi1        = afl((size_t)kB * 3 * kDH);
  float*  gh1        = afl((size_t)kB * 3 * kDH);
  float*  gi2        = afl((size_t)kB * 3 * kDH);
  float*  gh2        = afl((size_t)kB * 3 * kDH);
  float*  hsb        = afl((size_t)kB * kNATT);
  float*  logit      = afl((size_t)kB * kS);
  float*  alpha      = afl((size_t)kB * kS);
  float*  ctx        = afl((size_t)kB * kCTX);
  __bf16* ctx_bf     = abf((size_t)kB * kCTX);
  float*  ro         = afl((size_t)kB * kNRD);
  __bf16* ro_bf      = abf((size_t)kB * kNRD);
  float*  logits     = afl((size_t)kB * kV);
  float*  loss_acc   = afl((size_t)kB);

  auto conv = [&](const float* s, __bf16* d, size_t n) {
    rnnsearch_f2bf<<<dim3((unsigned)((n + 255) / 256)), 256, 0, stream>>>(s, d, (int)n);
  };
  auto gemm = [&](const __bf16* A, int lda, const __bf16* W, int ldw, const float* bias,
                  float* C, int ldc, int M, int N, int K, int flags) {
    rnnsearch_gemm_wmma<<<dim3(N / 128, M / 64), 128, 0, stream>>>(A, lda, W, ldw, bias, C, ldc, M, N, K, flags);
  };

  // ---- weight conversion (fp32 -> bf16) ----
  conv(P[2],  w_ef_ih, (size_t)3 * kEH * kEIN);
  conv(P[3],  w_ef_hh, (size_t)3 * kEH * kEH);
  conv(P[6],  w_eb_ih, (size_t)3 * kEH * kEIN);
  conv(P[7],  w_eb_hh, (size_t)3 * kEH * kEH);
  conv(P[10], w_g1_ih, (size_t)3 * kDH * 512);
  conv(P[11], w_g1_hh, (size_t)3 * kDH * kDH);
  conv(P[14], w_g2_ih, (size_t)3 * kDH * kCTX);
  conv(P[15], w_g2_hh, (size_t)3 * kDH * kDH);
  conv(P[18], w_h2s,   (size_t)kNATT * kDH);
  conv(P[20], w_s2s,   (size_t)kNATT * kCTX);
  conv(P[24], w_e2o,   (size_t)kNRD * 512);
  conv(P[26], w_h2o,   (size_t)kNRD * kDH);
  conv(P[28], w_c2o,   (size_t)kNRD * kCTX);
  conv(P[30], w_aff,   (size_t)kV * kNRD);
  conv(P[32], w_init,  (size_t)kDH * kCTX);

  // ---- embedding gathers ----
  rnnsearch_gather_src<<<dim3((kS * kB * kEIN) / 256), 256, 0, stream>>>(src, P[0], xs_bf);
  rnnsearch_gather_trg<<<dim3((kB * kT * 512) / 256), 256, 0, stream>>>(ftrg, P[1], temb_bf);

  // ---- encoder: hoisted x@Wih + bih for both directions (M = S*B = 4096) ----
  gemm(xs_bf, kEIN, w_ef_ih, kEIN, P[4], gi_fw, 3 * kEH, kS * kB, 3 * kEH, kEIN, GF_BIAS);
  gemm(xs_bf, kEIN, w_eb_ih, kEIN, P[8], gi_bk, 3 * kEH, kS * kB, 3 * kEH, kEIN, GF_BIAS);

  // forward scan
  rnnsearch_zero_state<<<dim3((kB * kEH) / 256), 256, 0, stream>>>(h_enc, h_enc_bf, kB * kEH);
  for (int s = 0; s < kS; ++s) {
    gemm(h_enc_bf, kEH, w_ef_hh, kEH, P[5], gh_enc, 3 * kEH, kB, 3 * kEH, kEH, GF_BIAS);
    rnnsearch_gru_enc_pw<<<dim3((kB * kEH) / 256), 256, 0, stream>>>(
        gi_fw + (size_t)s * kB * 3 * kEH, gh_enc, smask, s,
        h_enc, h_enc_bf, enc_ctx + (size_t)s * kCTX + 0, kS * kCTX);
  }
  // backward scan (process s descending; store at s, i.e. out_b[::-1])
  rnnsearch_zero_state<<<dim3((kB * kEH) / 256), 256, 0, stream>>>(h_enc, h_enc_bf, kB * kEH);
  for (int s = kS - 1; s >= 0; --s) {
    gemm(h_enc_bf, kEH, w_eb_hh, kEH, P[9], gh_enc, 3 * kEH, kB, 3 * kEH, kEH, GF_BIAS);
    rnnsearch_gru_enc_pw<<<dim3((kB * kEH) / 256), 256, 0, stream>>>(
        gi_bk + (size_t)s * kB * 3 * kEH, gh_enc, smask, s,
        h_enc, h_enc_bf, enc_ctx + (size_t)s * kCTX + kEH, kS * kCTX);
  }
  conv(enc_ctx, enc_ctx_bf, (size_t)kB * kS * kCTX);

  // ---- decoder init hidden + hoisted attention context projection ----
  rnnsearch_enc_avg<<<dim3((kB * kCTX) / 256), 256, 0, stream>>>(enc_ctx, smask, avg_bf);
  gemm(avg_bf, kCTX, w_init, kCTX, P[33], hid, kDH, kB, kDH, kCTX, GF_BIAS);
  rnnsearch_tanh_bf<<<dim3((kB * kDH) / 256), 256, 0, stream>>>(hid, hid_bf, kB * kDH);
  gemm(enc_ctx_bf, kCTX, w_s2s, kCTX, P[21], attn_ctx, kNATT, kB * kS, kNATT, kCTX, GF_BIAS);

  rnnsearch_init_out<<<1, 64, 0, stream>>>(loss_acc, out);

  // ---- decoder time loop (t = 0 .. T-2) ----
  for (int t = 0; t < kT - 1; ++t) {
    const __bf16* xt = temb_bf + (size_t)t * 512;            // lda = kT*512 (row = batch)
    // gru1
    gemm(xt, kT * 512, w_g1_ih, 512, P[12], gi1, 3 * kDH, kB, 3 * kDH, 512, GF_BIAS);
    gemm(hid_bf, kDH, w_g1_hh, kDH, P[13], gh1, 3 * kDH, kB, 3 * kDH, kDH, GF_BIAS);
    rnnsearch_gru_dec_pw<<<dim3((kB * kDH) / 256), 256, 0, stream>>>(gi1, gh1, hid, h1, h1_bf, kDH);
    // attention
    gemm(h1_bf, kDH, w_h2s, kDH, P[19], hsb, kNATT, kB, kNATT, kDH, GF_BIAS);
    rnnsearch_attn_logits<<<dim3((kB * kS) / 8), 256, 0, stream>>>(attn_ctx, hsb, P[22], P[23], smask, logit);
    rnnsearch_softmax<<<dim3(kB), 64, 0, stream>>>(logit, alpha);
    rnnsearch_attn_context<<<dim3((kB * kCTX) / 256), 256, 0, stream>>>(alpha, enc_ctx, ctx, ctx_bf);
    // gru2
    gemm(ctx_bf, kCTX, w_g2_ih, kCTX, P[16], gi2, 3 * kDH, kB, 3 * kDH, kCTX, GF_BIAS);
    gemm(h1_bf, kDH, w_g2_hh, kDH, P[17], gh2, 3 * kDH, kB, 3 * kDH, kDH, GF_BIAS);
    rnnsearch_gru_dec_pw<<<dim3((kB * kDH) / 256), 256, 0, stream>>>(gi2, gh2, h1, hid, hid_bf, kDH);
    // readout = tanh(x@e2o + h2@h2o + ctx@c2o + biases)
    gemm(xt, kT * 512, w_e2o, 512, P[25], ro, kNRD, kB, kNRD, 512, GF_BIAS);
    gemm(hid_bf, kDH, w_h2o, kDH, P[27], ro, kNRD, kB, kNRD, kDH, GF_BIAS | GF_ACC);
    gemm(ctx_bf, kCTX, w_c2o, kCTX, P[29], ro, kNRD, kB, kNRD, kCTX, GF_BIAS | GF_ACC);
    rnnsearch_tanh_bf<<<dim3((kB * kNRD) / 256), 256, 0, stream>>>(ro, ro_bf, kB * kNRD);
    // vocab projection + loss/argmax
    gemm(ro_bf, kNRD, w_aff, kNRD, P[31], logits, kV, kB, kV, kNRD, GF_BIAS);
    rnnsearch_vocab_loss<<<dim3(kB), 256, 0, stream>>>(logits, ftrg, tmask, t, loss_acc, out + 2);
  }

  rnnsearch_finalize<<<1, 64, 0, stream>>>(loss_acc, tmask, out);
}
